// TreeLSTMWithNodeSupervision_82703890252317
// MI455X (gfx1250) — matrix operand order repaired and checked by reading, not compiled, not run on hardware
//
#include <hip/hip_runtime.h>
#include <hip/hip_bf16.h>

// ---------------------------------------------------------------------------
// SPINN shift-reduce TreeLSTM for MI455X (gfx1250, wave32, WMMA).
// B=128, T=256, E=300, H=256, V=32000, O=5.
//
//  k_pack : fp32 weights -> bf16 packed  (Wc=[Wp;Wg] padded K->320, Wr)
//  k_proj : buffers[b,t,0:256]=h, [256:512]=c via v_wmma_f32_16x16x32_bf16
//  k_rnn  : 8 WGs x 16 batch rows, 511 lockstep steps; reduce cell is a
//           [16,512]x[512,1280] bf16 WMMA GEMM; gates combined in-register.
//           sched_group_barrier pins the per-k-chunk shape to
//           {20 VMEM_READ, 2 DS_READ, 10 WMMA} so loads stay in flight
//           beneath the WMMA burst (low occupancy -> ILP must hide latency).
//  k_out  : logits = final_h @ Wo^T + bo
// ---------------------------------------------------------------------------

typedef __attribute__((ext_vector_type(16))) __bf16 v16bf;
typedef __attribute__((ext_vector_type(8)))  __bf16 v8bf;
typedef __attribute__((ext_vector_type(8)))  float  v8f;

#define Bn    128
#define Tn    256
#define En    300
#define EP    320          // E padded to K-multiple of 32
#define Hn    256
#define TWOH  512
#define FIVEH 1280
#define NSTEP 511          // 2T-1
#define On    5

// workspace layout (bytes)
#define OFF_WC   ((size_t)0)                         // bf16 [512][320]
#define OFF_WR   ((size_t)327680)                    // bf16 [1280][512]
#define OFF_SPTR ((size_t)1638400)                   // int  [128]
#define OFF_BUF  ((size_t)2097152)                   // f32  [B*T][512]
#define OFF_STK  (OFF_BUF + (size_t)Bn*Tn*TWOH*4)    // f32  [B][T][512]

#if __has_builtin(__builtin_amdgcn_sched_group_barrier)
#define SGB(mask, size, sync) __builtin_amdgcn_sched_group_barrier((mask), (size), (sync))
#else
#define SGB(mask, size, sync)
#endif
#define SGB_VMEM_RD 0x020
#define SGB_DS_RD   0x100
#define SGB_MFMA    0x008

__device__ __forceinline__ float sigf(float x) { return 1.0f / (1.0f + __expf(-x)); }

// A fragment: 16x32 bf16 tile from LDS, contiguous K order (matches B):
// lane l: row = l&15, khalf = l>>4; elems 0..15 -> K = k0+khalf*16..
template <int STRIDE>
__device__ __forceinline__ v16bf load_a_frag(const __bf16 (*As)[STRIDE], int row, int khalf, int k0) {
    const v8bf lo = *(const v8bf*)&As[row][k0 + khalf * 16];
    const v8bf hi = *(const v8bf*)&As[row][k0 + khalf * 16 + 8];
    v16bf a;
#pragma unroll
    for (int j = 0; j < 8; ++j) { a[j] = lo[j]; a[j + 8] = hi[j]; }
    return a;
}

// ---------------------------------------------------------------------------
// k_pack: fp32 -> bf16 weight packing
// ---------------------------------------------------------------------------
__global__ __launch_bounds__(256) void k_pack(const float* __restrict__ Wp,
                                              const float* __restrict__ Wg,
                                              const float* __restrict__ Wr,
                                              __bf16* __restrict__ Wc,
                                              __bf16* __restrict__ Wrb) {
    int i = blockIdx.x * 256 + threadIdx.x;
    if (i < 512 * EP) {
        int r = i / EP, e = i % EP;
        float v = 0.0f;
        if (e < En) v = (r < Hn) ? Wp[r * En + e] : Wg[(r - Hn) * En + e];
        Wc[i] = (__bf16)v;
    }
    if (i < FIVEH * TWOH) Wrb[i] = (__bf16)Wr[i];
}

// ---------------------------------------------------------------------------
// k_proj: embedding gather + [16,320]x[320,512] WMMA, fused gate epilogue.
// Wave w owns h-tiles {w, w+8} and for each BOTH the c-column (n=h) and
// g-column (n=256+h) accumulators, so tanh*sigmoid fuses in registers.
// ---------------------------------------------------------------------------
__global__ __launch_bounds__(256) void k_proj(const int* __restrict__ x_ids,
                                              const float* __restrict__ embed,
                                              const float* __restrict__ bp,
                                              const float* __restrict__ bg,
                                              const __bf16* __restrict__ Wc,
                                              float* __restrict__ buffers) {
    __shared__ __bf16 As[16][EP];
    __shared__ int s_id[16];

    const int tid = threadIdx.x, lane = tid & 31, wave = tid >> 5;
    const int ft0 = blockIdx.x * 16;

    if (tid < 16) s_id[tid] = x_ids[ft0 + tid];
    __syncthreads();

    for (int i = tid; i < 16 * EP; i += 256) {
        int m = i / EP, e = i % EP;
        As[m][e] = (e < En) ? (__bf16)embed[(size_t)s_id[m] * En + e] : (__bf16)0.0f;
    }
    __syncthreads();

    v8f acc[4] = {};   // u = t*2 + {0:c, 1:g}
    const int col = lane & 15, khalf = lane >> 4;
    // single per-lane base index; fragment rows differ by compile-time consts
    const size_t base = (size_t)(wave * 16 + col) * EP + khalf * 16;

    for (int kc = 0; kc < EP / 32; ++kc) {
        const int k0 = kc * 32;
        v16bf bf[4];
#pragma unroll
        for (int t = 0; t < 2; ++t) {
#pragma unroll
            for (int g = 0; g < 2; ++g)   // g=0: c rows, g=1: g rows (+Hn)
                bf[t * 2 + g] = *(const v16bf*)&Wc[base + k0 + (size_t)(t * 128 + g * Hn) * EP];
        }
        const v16bf a = load_a_frag<EP>(As, col, khalf, k0);
        SGB(SGB_VMEM_RD, 8, 0);   // 8x global_load_b128 (4 fragments)
        SGB(SGB_DS_RD,   2, 0);   // A fragment
        SGB(SGB_MFMA,    4, 0);   // 4 WMMAs
#pragma unroll
        for (int u = 0; u < 4; ++u)
            acc[u] = __builtin_amdgcn_wmma_f32_16x16x32_bf16(false, a, false, bf[u],
                                                             (short)0, acc[u], false, false);
    }

#pragma unroll
    for (int t = 0; t < 2; ++t) {
        const int h = (wave + t * 8) * 16 + col;
        const float bpv = bp[h], bgv = bg[h];
#pragma unroll
        for (int v = 0; v < 8; ++v) {
            const int m = v + 8 * khalf;            // D layout: row = v + 8*(lane>>4)
            const float c = acc[t * 2 + 0][v] + bpv;
            const float g = acc[t * 2 + 1][v] + bgv;
            float* row = buffers + (size_t)(ft0 + m) * TWOH;
            row[h]      = tanhf(c) * sigf(g);       // h_buf
            row[Hn + h] = c;                        // c_buf
        }
    }
}

// ---------------------------------------------------------------------------
// k_rnn: 8 workgroups x 16 batch elements, 511 lockstep shift/reduce steps.
// ---------------------------------------------------------------------------
__global__ __launch_bounds__(256) void k_rnn(const int* __restrict__ trans,    // [511,128]
                                             const float* __restrict__ buffers,
                                             const float* __restrict__ br,
                                             const __bf16* __restrict__ Wr,    // bf16 [1280][512]
                                             float* __restrict__ stack,        // [128][256][512]
                                             int* __restrict__ sptrF) {
    __shared__ __bf16 As[16][TWOH];          // [m=16 batch rows][hl|hr]
    __shared__ float s_br[FIVEH];
    __shared__ int s_sptr[16], s_bptr[16], s_trans[16];

    const int tid = threadIdx.x, lane = tid & 31, wave = tid >> 5;
    const int b0 = blockIdx.x * 16;
    const int col = lane & 15, khalf = lane >> 4;
    // u = t*5 + g maps to Wr row (g*256 + t*128 + wave*16 + col)
    const size_t base = (size_t)(wave * 16 + col) * TWOH + khalf * 16;

    for (int i = tid; i < FIVEH; i += 256) s_br[i] = br[i];
    if (tid < 16) { s_sptr[tid] = 0; s_bptr[tid] = Tn; }
    __syncthreads();

    for (int step = 0; step < NSTEP; ++step) {
        if (tid < 16) s_trans[tid] = trans[step * Bn + b0 + tid];
        __syncthreads();

        // ---- phase A: shift copies + stage reduce operands (hl|hr) in LDS
        for (int i = tid; i < 16 * TWOH; i += 256) {
            const int m = i >> 9, k = i & 511;
            const int sp = s_sptr[m];
            float* srow = stack + (size_t)(b0 + m) * Tn * TWOH;
            if (s_trans[m] == 0) {   // SHIFT: push buffers[b, bptr-1] at slot sptr
                srow[(long)sp * TWOH + k] =
                    buffers[((size_t)(b0 + m) * Tn + (s_bptr[m] - 1)) * TWOH + k];
            } else {                 // REDUCE: A row = [hl | hr]
                const int slot = (k < Hn) ? sp - 2 : sp - 1;
                As[m][k] = (__bf16)srow[(long)slot * TWOH + (k & 255)];
            }
        }
        __syncthreads();

        // ---- phase B: proj[16,1280] = A @ Wr^T.
        // Scheduler pinned to {20 VMEM_READ, 2 DS_READ, 10 WMMA} per k-chunk:
        // in-order load completion means WMMA u waits only for its own loads
        // while later fragments remain in flight.
        v8f acc[10] = {};
        for (int kc = 0; kc < TWOH / 32; ++kc) {
            const int k0 = kc * 32;
            v16bf bf[10];
#pragma unroll
            for (int t = 0; t < 2; ++t) {
#pragma unroll
                for (int g = 0; g < 5; ++g)
                    bf[t * 5 + g] =
                        *(const v16bf*)&Wr[base + k0 + (size_t)(g * 256 + t * 128) * TWOH];
            }
            const v16bf a = load_a_frag<TWOH>(As, col, khalf, k0);
            SGB(SGB_VMEM_RD, 20, 0);   // 20x global_load_b128 (10 fragments)
            SGB(SGB_DS_RD,    2, 0);   // A fragment
            SGB(SGB_MFMA,    10, 0);   // 10 WMMAs
#pragma unroll
            for (int u = 0; u < 10; ++u)
                acc[u] = __builtin_amdgcn_wmma_f32_16x16x32_bf16(false, a, false, bf[u],
                                                                 (short)0, acc[u], false, false);
        }

        // ---- phase C: in-register gate math, write (h_new,c_new) for reduce rows
#pragma unroll
        for (int t = 0; t < 2; ++t) {
            const int h = (wave + t * 8) * 16 + col;
            const float bI = s_br[h],          bFl = s_br[Hn + h], bFr = s_br[2 * Hn + h];
            const float bG = s_br[3 * Hn + h], bO  = s_br[4 * Hn + h];
#pragma unroll
            for (int v = 0; v < 8; ++v) {
                const int m = v + 8 * khalf;
                if (s_trans[m] == 1) {
                    const int sp = s_sptr[m];
                    float* srow = stack + (size_t)(b0 + m) * Tn * TWOH;
                    const float cl = srow[(long)(sp - 2) * TWOH + Hn + h];
                    const float cr = srow[(long)(sp - 1) * TWOH + Hn + h];
                    const float iv = acc[t * 5 + 0][v] + bI;
                    const float fl = acc[t * 5 + 1][v] + bFl;
                    const float fr = acc[t * 5 + 2][v] + bFr;
                    const float gv = acc[t * 5 + 3][v] + bG;
                    const float ov = acc[t * 5 + 4][v] + bO;
                    const float cn = sigf(iv) * tanhf(gv) + sigf(fl) * cl + sigf(fr) * cr;
                    srow[(long)(sp - 2) * TWOH + h]      = sigf(ov) * tanhf(cn);
                    srow[(long)(sp - 2) * TWOH + Hn + h] = cn;
                }
            }
        }
        __threadfence_block();
        __syncthreads();

        if (tid < 16) {
            if (s_trans[tid] == 0) { s_sptr[tid] += 1; s_bptr[tid] -= 1; }
            else                   { s_sptr[tid] -= 1; }
        }
        __syncthreads();
    }
    if (tid < 16) sptrF[b0 + tid] = s_sptr[tid];
}

// ---------------------------------------------------------------------------
// k_out: logits[b,o] = dot(stack[b, sptr-1, 0:256], Wo[o]) + bo[o]
// ---------------------------------------------------------------------------
__global__ __launch_bounds__(256) void k_out(const float* __restrict__ stack,
                                             const int* __restrict__ sptrF,
                                             const float* __restrict__ Wo,
                                             const float* __restrict__ bo,
                                             float* __restrict__ out) {
    int i = blockIdx.x * 256 + threadIdx.x;
    if (i >= Bn * On) return;
    const int b = i / On, o = i % On;
    const float* hrow = stack + ((size_t)b * Tn + (sptrF[b] - 1)) * TWOH;
    float s = bo[o];
    for (int h = 0; h < Hn; ++h) s += hrow[h] * Wo[o * Hn + h];
    out[i] = s;
}

// ---------------------------------------------------------------------------
extern "C" void kernel_launch(void* const* d_in, const int* in_sizes, int n_in,
                              void* d_out, int out_size, void* d_ws, size_t ws_size,
                              hipStream_t stream) {
    const int*   x_ids = (const int*)d_in[0];
    const int*   trans = (const int*)d_in[1];
    const float* embed = (const float*)d_in[2];
    const float* Wp    = (const float*)d_in[3];
    const float* bp    = (const float*)d_in[4];
    const float* Wg    = (const float*)d_in[5];
    const float* bg    = (const float*)d_in[6];
    const float* Wr    = (const float*)d_in[7];
    const float* br    = (const float*)d_in[8];
    const float* Wo    = (const float*)d_in[9];
    const float* bo    = (const float*)d_in[10];

    char* ws = (char*)d_ws;
    __bf16* Wc      = (__bf16*)(ws + OFF_WC);
    __bf16* Wrb     = (__bf16*)(ws + OFF_WR);
    int*    sptrF   = (int*)(ws + OFF_SPTR);
    float*  buffers = (float*)(ws + OFF_BUF);
    float*  stack   = (float*)(ws + OFF_STK);

    k_pack<<<(FIVEH * TWOH + 255) / 256, 256, 0, stream>>>(Wp, Wg, Wr, Wc, Wrb);
    k_proj<<<(Bn * Tn) / 16, 256, 0, stream>>>(x_ids, embed, bp, bg, Wc, buffers);
    k_rnn<<<Bn / 16, 256, 0, stream>>>(trans, buffers, br, Wrb, stack, sptrF);
    k_out<<<(Bn * On + 255) / 256, 256, 0, stream>>>(stack, sptrF, Wo, bo, (float*)d_out);
}
